// EvolvedLoopLinear_21251498180730
// MI455X (gfx1250) — matrix operand order repaired and checked by reading, not compiled, not run on hardware
//
#include <hip/hip_runtime.h>

typedef __attribute__((ext_vector_type(16))) __bf16 v16bf;
typedef __attribute__((ext_vector_type(8)))  float  v8f;
typedef __attribute__((ext_vector_type(4)))  float  v4f;
typedef __attribute__((ext_vector_type(4)))  __bf16 v4bf;
typedef __attribute__((ext_vector_type(4)))  int    v4i;

#define M_DIM 4096
#define N_DIM 4096
#define K_DIM 4096

union Frag16 { v16bf v; uint4 q[2]; };

#if __has_builtin(__builtin_amdgcn_global_load_async_to_lds_b128) && \
    __has_builtin(__builtin_amdgcn_s_wait_asynccnt)
#define USE_ASYNC 1
#endif

#ifdef USE_ASYNC
// per compile-probe diagnostic: param0 = v4i in addrspace(1) (__device__),
// param1 = LDS-side pointer (addrspace(3)), then imm offset + imm cpol
typedef __attribute__((address_space(1))) v4i as1_v4i;
typedef __attribute__((address_space(3))) v4i as3_v4i;
__device__ __forceinline__ void async_cp16(const unsigned short* gp, unsigned short* lp) {
  __builtin_amdgcn_global_load_async_to_lds_b128((as1_v4i*)gp, (as3_v4i*)lp, 0, 0);
}
#endif

// ---------------------------------------------------------------------------
// Pass 1: one-shot f32 -> bf16 conversion (RNE via fptrunc / v_cvt_pk_bf16)
// ---------------------------------------------------------------------------
__global__ __launch_bounds__(256) void cvt_f32_to_bf16(
    const float* __restrict__ src, unsigned short* __restrict__ dst, int n4) {
  const int stride = (int)(gridDim.x * blockDim.x);
  for (int i = (int)(blockIdx.x * blockDim.x + threadIdx.x); i < n4; i += stride) {
    const float4 f = ((const float4*)src)[i];
    v4f vf; vf[0] = f.x; vf[1] = f.y; vf[2] = f.z; vf[3] = f.w;
    const v4bf c = __builtin_convertvector(vf, v4bf);
    *(v4bf*)(dst + (size_t)i * 4) = c;   // 8-byte store
  }
}

// ---------------------------------------------------------------------------
// Pass 2: bf16 GEMM  out = xb @ wb^T + bias   (f32 accumulate)
//   block tile 128x128, K-tile 64, 8 waves, wave tile 64x32 (4x2 WMMA 16x16x32)
//   async global->LDS double-buffered staging when available
// ---------------------------------------------------------------------------
#define BM2 128
#define BN2 128
#define BK2 64
#define LDT 72          /* LDS row stride in bf16 (144B: conflict-free b128) */
#define KTILES (K_DIM / BK2)

__global__ __launch_bounds__(256) void gemm_bf16_wmma(
    const unsigned short* __restrict__ xb,
    const unsigned short* __restrict__ wb,
    const float* __restrict__ bias,
    float* __restrict__ out) {
#ifdef USE_ASYNC
  __shared__ unsigned short As[2][BM2 * LDT];
  __shared__ unsigned short Bs[2][BN2 * LDT];
#else
  __shared__ unsigned short As[1][BM2 * LDT];
  __shared__ unsigned short Bs[1][BN2 * LDT];
#endif

  const int tid   = threadIdx.x;
  const int lane  = tid & 31;        // wave32
  const int wave  = tid >> 5;        // 0..7
  const int waveM = wave >> 2;       // 0..1  (64 rows each)
  const int waveN = wave & 3;        // 0..3  (32 cols each)
  const int l16   = lane & 15;
  const int half  = lane >> 4;

  const int blockRow = blockIdx.y * BM2;
  const int blockCol = blockIdx.x * BN2;

  // accumulators pre-loaded with bias (broadcast along M)
  v8f acc[4][2];
#pragma unroll
  for (int n = 0; n < 2; ++n) {
    const float bv = bias[blockCol + waveN * 32 + n * 16 + l16];
#pragma unroll
    for (int m = 0; m < 4; ++m)
#pragma unroll
      for (int r = 0; r < 8; ++r) acc[m][n][r] = bv;
  }

#ifdef USE_ASYNC
  auto stage = [&](int buf, int kt) {
#pragma unroll
    for (int i = 0; i < 4; ++i) {
      const int idx = tid + i * 256;
      const int row = idx >> 3;      // 0..127
      const int c   = idx & 7;       // 16B chunk within 128B row
      async_cp16(xb + (size_t)(blockRow + row) * K_DIM + kt * BK2 + c * 8,
                 &As[buf][row * LDT + c * 8]);
      async_cp16(wb + (size_t)(blockCol + row) * K_DIM + kt * BK2 + c * 8,
                 &Bs[buf][row * LDT + c * 8]);
    }
  };
#else
  auto stage = [&](int buf, int kt) {
#pragma unroll
    for (int i = 0; i < 4; ++i) {
      const int idx = tid + i * 256;
      const int row = idx >> 3;
      const int c   = idx & 7;
      const uint4 va = *(const uint4*)(xb + (size_t)(blockRow + row) * K_DIM + kt * BK2 + c * 8);
      const uint4 vb = *(const uint4*)(wb + (size_t)(blockCol + row) * K_DIM + kt * BK2 + c * 8);
      *(uint4*)&As[buf][row * LDT + c * 8] = va;
      *(uint4*)&Bs[buf][row * LDT + c * 8] = vb;
    }
  };
#endif

  auto compute = [&](int buf) {
#pragma unroll
    for (int ks = 0; ks < 2; ++ks) {           // two K=32 sub-steps of BK2=64
      const int kk = ks * 32;
      Frag16 a[4], bfr[2];
#pragma unroll
      for (int m = 0; m < 4; ++m) {
        // A 16x32 bf16: lanes 0-15 hold K0..7 & K16..23; lanes 16-31 K8..15 & K24..31
        const unsigned short* ap = &As[buf][(waveM * 64 + m * 16 + l16) * LDT + kk];
        a[m].q[0] = *(const uint4*)(ap + half * 8);
        a[m].q[1] = *(const uint4*)(ap + 16 + half * 8);
      }
#pragma unroll
      for (int n = 0; n < 2; ++n) {
        // B 32x16 bf16: lane = column N=l16; lanes 0-15 hold K0..7 & K16..23; lanes 16-31 K8..15 & K24..31
        const unsigned short* bp = &Bs[buf][(waveN * 32 + n * 16 + l16) * LDT + kk];
        bfr[n].q[0] = *(const uint4*)(bp + half * 8);
        bfr[n].q[1] = *(const uint4*)(bp + 16 + half * 8);
      }
#pragma unroll
      for (int m = 0; m < 4; ++m)
#pragma unroll
        for (int n = 0; n < 2; ++n)
          acc[m][n] = __builtin_amdgcn_wmma_f32_16x16x32_bf16(
              false, a[m].v, false, bfr[n].v, (short)0, acc[m][n], false, false);
    }
  };

#ifdef USE_ASYNC
  stage(0, 0);
  for (int t = 0; t < KTILES; ++t) {
    const int buf = t & 1;
    if (t + 1 < KTILES) {
      stage(buf ^ 1, t + 1);                     // overlap next tile's DMA
      __builtin_amdgcn_s_wait_asynccnt(8);       // this tile's 8 copies done
    } else {
      __builtin_amdgcn_s_wait_asynccnt(0);
    }
    __syncthreads();
    compute(buf);
    __syncthreads();                             // protect buffer reuse
  }
#else
  for (int t = 0; t < KTILES; ++t) {
    stage(0, t);
    __syncthreads();
    compute(0);
    __syncthreads();
  }
#endif

  // store D: VGPR r -> row half*8 + r, col l16
#pragma unroll
  for (int m = 0; m < 4; ++m)
#pragma unroll
    for (int n = 0; n < 2; ++n)
#pragma unroll
      for (int r = 0; r < 8; ++r) {
        const int row = blockRow + waveM * 64 + m * 16 + half * 8 + r;
        const int col = blockCol + waveN * 32 + n * 16 + l16;
        out[(size_t)row * N_DIM + col] = acc[m][n][r];
      }
}

// ---------------------------------------------------------------------------
// Fallback: fused f32->bf16-in-LDS kernel (round-1 style) if d_ws too small
// ---------------------------------------------------------------------------
#define BM1 128
#define BN1 64
#define BK1 32
#define LD1 40

__device__ __forceinline__ unsigned pack2_bf16(float lo, float hi) {
  unsigned a = __float_as_uint(lo), b = __float_as_uint(hi);
  a = (a + 0x7FFFu + ((a >> 16) & 1u)) >> 16;
  b = (b + 0x7FFFu + ((b >> 16) & 1u)) >> 16;
  return (b << 16) | (a & 0xFFFFu);
}

__global__ __launch_bounds__(256) void linear_fused_fallback(
    const float* __restrict__ x, const float* __restrict__ w,
    const float* __restrict__ bias, float* __restrict__ out) {
  __shared__ unsigned short As[BM1 * LD1];
  __shared__ unsigned short Bs[BN1 * LD1];
  const int tid = threadIdx.x, wave = tid >> 5, lane = tid & 31;
  const int waveM = wave >> 1, waveN = wave & 1;
  const int l16 = lane & 15, half = lane >> 4;
  const int blockRow = blockIdx.y * BM1, blockCol = blockIdx.x * BN1;

  v8f acc[2][2];
#pragma unroll
  for (int j = 0; j < 2; ++j) {
    const float bv = bias[blockCol + waveN * 32 + j * 16 + l16];
#pragma unroll
    for (int i = 0; i < 2; ++i)
#pragma unroll
      for (int r = 0; r < 8; ++r) acc[i][j][r] = bv;
  }
  for (int k0 = 0; k0 < K_DIM; k0 += BK1) {
#pragma unroll
    for (int i = 0; i < 4; ++i) {
      const int idx = tid + i * 256, row = idx >> 3, c4 = idx & 7;
      const float4 f = *(const float4*)(x + (size_t)(blockRow + row) * K_DIM + k0 + c4 * 4);
      uint2 p; p.x = pack2_bf16(f.x, f.y); p.y = pack2_bf16(f.z, f.w);
      *(uint2*)&As[row * LD1 + c4 * 4] = p;
    }
#pragma unroll
    for (int i = 0; i < 2; ++i) {
      const int idx = tid + i * 256, row = idx >> 3, c4 = idx & 7;
      const float4 f = *(const float4*)(w + (size_t)(blockCol + row) * K_DIM + k0 + c4 * 4);
      uint2 p; p.x = pack2_bf16(f.x, f.y); p.y = pack2_bf16(f.z, f.w);
      *(uint2*)&Bs[row * LD1 + c4 * 4] = p;
    }
    __syncthreads();
    Frag16 a[2], bfr[2];
#pragma unroll
    for (int i = 0; i < 2; ++i) {
      const unsigned short* ap = &As[(waveM * 32 + i * 16 + l16) * LD1];
      a[i].q[0] = *(const uint4*)(ap + half * 8);
      a[i].q[1] = *(const uint4*)(ap + 16 + half * 8);
    }
#pragma unroll
    for (int j = 0; j < 2; ++j) {
      const unsigned short* bp = &Bs[(waveN * 32 + j * 16 + l16) * LD1];
      bfr[j].q[0] = *(const uint4*)(bp + half * 8);
      bfr[j].q[1] = *(const uint4*)(bp + 16 + half * 8);
    }
#pragma unroll
    for (int i = 0; i < 2; ++i)
#pragma unroll
      for (int j = 0; j < 2; ++j)
        acc[i][j] = __builtin_amdgcn_wmma_f32_16x16x32_bf16(
            false, a[i].v, false, bfr[j].v, (short)0, acc[i][j], false, false);
    __syncthreads();
  }
#pragma unroll
  for (int i = 0; i < 2; ++i)
#pragma unroll
    for (int j = 0; j < 2; ++j)
#pragma unroll
      for (int r = 0; r < 8; ++r) {
        const int row = blockRow + waveM * 32 + i * 16 + half * 8 + r;
        const int col = blockCol + waveN * 32 + j * 16 + l16;
        out[(size_t)row * N_DIM + col] = acc[i][j][r];
      }
}

// ---------------------------------------------------------------------------
extern "C" void kernel_launch(void* const* d_in, const int* in_sizes, int n_in,
                              void* d_out, int out_size, void* d_ws, size_t ws_size,
                              hipStream_t stream) {
  (void)in_sizes; (void)n_in; (void)out_size;
  const float* x = (const float*)d_in[0];
  const float* w = (const float*)d_in[1];
  const float* b = (const float*)d_in[2];
  float* out = (float*)d_out;

  const size_t elems = (size_t)M_DIM * K_DIM;              // 16M per matrix
  const size_t need  = 2 * elems * sizeof(unsigned short); // 64 MB

  if (ws_size >= need) {
    unsigned short* xb = (unsigned short*)d_ws;
    unsigned short* wbp = xb + elems;
    hipLaunchKernelGGL(cvt_f32_to_bf16, dim3(2048), dim3(256), 0, stream,
                       x, xb, (int)(elems / 4));
    hipLaunchKernelGGL(cvt_f32_to_bf16, dim3(2048), dim3(256), 0, stream,
                       w, wbp, (int)(elems / 4));
    hipLaunchKernelGGL(gemm_bf16_wmma, dim3(N_DIM / BN2, M_DIM / BM2), dim3(256),
                       0, stream, xb, wbp, b, out);
  } else {
    hipLaunchKernelGGL(linear_fused_fallback, dim3(N_DIM / BN1, M_DIM / BM1),
                       dim3(256), 0, stream, x, w, b, out);
  }
}
